// Attention_42477226557744
// MI455X (gfx1250) — matrix operand order repaired and checked by reading, not compiled, not run on hardware
//
#include <hip/hip_runtime.h>
#include <cstdint>

#define DEV __device__ __forceinline__

typedef __attribute__((ext_vector_type(16))) __bf16 v16bf;
typedef __attribute__((ext_vector_type(8)))  float  v8f;
typedef __attribute__((ext_vector_type(4)))  unsigned int v4u;
typedef __attribute__((ext_vector_type(8)))  int v8i;
typedef __attribute__((ext_vector_type(4)))  int v4i;

union Frag16 { uint4 u[2]; v16bf v; };

DEV unsigned short f2bf(float x) {
  union { float f; unsigned u; } v; v.f = x;
  unsigned r = v.u + 0x7FFFu + ((v.u >> 16) & 1u);  // round-to-nearest-even
  return (unsigned short)(r >> 16);
}

// A-fragment (16xK tile, row = lane%16): lanes 0-15 hold K 0..7 & 16..23,
// lanes 16-31 hold K 8..15 & 24..31 (ISA 16-bit A layout).
DEV v16bf load_fragA(const unsigned short* rowPtr, int lane) {
  const char* p = (const char*)rowPtr;
  int sel = (lane & 16) ? 16 : 0;
  Frag16 f;
  f.u[0] = *(const uint4*)(p + sel);
  f.u[1] = *(const uint4*)(p + 32 + sel);
  return f.v;
}
// B-fragment (Kx16 tile, col = lane%16): lanes 0-15 hold K 0..7 & 16..23,
// lanes 16-31 hold K 8..15 & 24..31 (matches A layout).
DEV v16bf load_fragB(const unsigned short* colPtr, int lane) {
  const char* p = (const char*)colPtr;
  int sel = (lane & 16) ? 16 : 0;
  Frag16 f;
  f.u[0] = *(const uint4*)(p + sel);
  f.u[1] = *(const uint4*)(p + 32 + sel);
  return f.v;
}

DEV v8f wmma_bf16(v16bf a, v16bf b, v8f c) {
  return __builtin_amdgcn_wmma_f32_16x16x32_bf16(false, a, false, b, (short)0, c,
                                                 false, false);
}

DEV float wave_sum32(float v) {
  #pragma unroll
  for (int m = 1; m < 32; m <<= 1) v += __shfl_xor(v, m, 32);
  return v;
}

DEV ushort4 pack4(float4 v) {
  return make_ushort4(f2bf(v.x), f2bf(v.y), f2bf(v.z), f2bf(v.w));
}

// Async copy of one 16B chunk per call: global -> LDS, ASYNCcnt-tracked.
DEV void async_b128(unsigned ldsAddr, const void* gAddr) {
  asm volatile("global_load_async_to_lds_b128 %0, %1, off"
               :: "v"(ldsAddr), "v"((unsigned long long)(uintptr_t)gAddr)
               : "memory");
}

// Copy a 128x32-halves (8KB) tile: 256 threads x 2 chunks of 16B.
DEV void async_cp_tile(const unsigned short* gRow0, int strideHalves,
                       unsigned ldsBase, int tid) {
  int r = tid >> 2;
  int cb = (tid & 3) * 16;  // byte offset within 64B row
  async_b128(ldsBase + (unsigned)(r * 64 + cb),
             (const char*)(gRow0 + (size_t)r * strideHalves) + cb);
  async_b128(ldsBase + (unsigned)((r + 64) * 64 + cb),
             (const char*)(gRow0 + (size_t)(r + 64) * strideHalves) + cb);
}

// ---------------------------------------------------------------- cvt / transpose
__global__ __launch_bounds__(256) void k_cvt_bf16(const float* __restrict__ src,
                                                  unsigned short* __restrict__ dst,
                                                  int n) {
  int i = blockIdx.x * 256 + threadIdx.x;
  if (i < n) dst[i] = f2bf(src[i]);
}

// WT[n*K + k] = W[k*N + n]  (bf16)
__global__ __launch_bounds__(256) void k_transpose_cvt(const float* __restrict__ W,
                                                       unsigned short* __restrict__ WT,
                                                       int K, int N) {
  int i = blockIdx.x * 256 + threadIdx.x;
  if (i < K * N) {
    int k = i % K, n = i / K;
    WT[i] = f2bf(W[(size_t)k * N + n]);
  }
}

// ---------------------------------------------------------------- GEMM (bf16 -> f32)
// C[M,N] = A[M,K] @ B^T (Bm stored [N][K]). 128x128 tile, 8 waves, K step 32.
// Double-buffered LDS staging via global_load_async_to_lds_b128 (ASYNCcnt).
__global__ __launch_bounds__(256) void k_gemm_bf16(const unsigned short* __restrict__ A,
                                                   const unsigned short* __restrict__ Bm,
                                                   float* __restrict__ C,
                                                   int M, int N, int K) {
  __shared__ unsigned short Abuf[2][128 * 32];
  __shared__ unsigned short Bbuf[2][128 * 32];
  int tid = threadIdx.x;
  int lane = tid & 31, wave = tid >> 5;
  int lo = lane & 15, grp = lane >> 4;
  int m_base = blockIdx.y * 128;
  int n_base = blockIdx.x * 128;
  int wm = (wave & 3) * 32, wn = (wave >> 2) * 64;

  unsigned aB[2] = {(unsigned)(size_t)(void*)&Abuf[0][0],
                    (unsigned)(size_t)(void*)&Abuf[1][0]};
  unsigned bB[2] = {(unsigned)(size_t)(void*)&Bbuf[0][0],
                    (unsigned)(size_t)(void*)&Bbuf[1][0]};
  const unsigned short* Ag = A + (size_t)m_base * K;
  const unsigned short* Bg = Bm + (size_t)n_base * K;

  v8f z = {0.f, 0.f, 0.f, 0.f, 0.f, 0.f, 0.f, 0.f};
  v8f acc[2][4];
  #pragma unroll
  for (int mt = 0; mt < 2; ++mt)
    #pragma unroll
    for (int nt = 0; nt < 4; ++nt) acc[mt][nt] = z;

  async_cp_tile(Ag, K, aB[0], tid);
  async_cp_tile(Bg, K, bB[0], tid);

  int nk = K / 32;
  for (int i = 0; i < nk; ++i) {
    int cur = i & 1;
    if (i + 1 < nk) {
      async_cp_tile(Ag + (size_t)(i + 1) * 32, K, aB[cur ^ 1], tid);
      async_cp_tile(Bg + (size_t)(i + 1) * 32, K, bB[cur ^ 1], tid);
      asm volatile("s_wait_asynccnt 4" ::: "memory");  // current tile landed
    } else {
      asm volatile("s_wait_asynccnt 0" ::: "memory");
    }
    __syncthreads();

    const unsigned short* As = (const unsigned short*)&Abuf[cur][0];
    const unsigned short* Bs = (const unsigned short*)&Bbuf[cur][0];
    v16bf af[2], bfr[4];
    #pragma unroll
    for (int mt = 0; mt < 2; ++mt)
      af[mt] = load_fragA(As + (size_t)(wm + mt * 16 + lo) * 32, lane);
    #pragma unroll
    for (int nt = 0; nt < 4; ++nt)
      bfr[nt] = load_fragB(Bs + (size_t)(wn + nt * 16 + lo) * 32, lane);
    #pragma unroll
    for (int mt = 0; mt < 2; ++mt)
      #pragma unroll
      for (int nt = 0; nt < 4; ++nt)
        acc[mt][nt] = wmma_bf16(af[mt], bfr[nt], acc[mt][nt]);
    __syncthreads();
  }

  #pragma unroll
  for (int mt = 0; mt < 2; ++mt)
    #pragma unroll
    for (int nt = 0; nt < 4; ++nt)
      #pragma unroll
      for (int e = 0; e < 8; ++e)
        C[(size_t)(m_base + wm + mt * 16 + grp * 8 + e) * N +
          n_base + wn + nt * 16 + lo] = acc[mt][nt][e];
}

// ---------------------------------------------------------------- l2norm/scale/RoPE
DEV float4 rope4(float4 t, const float* __restrict__ freqRow, int lane) {
  float4 tp;
  tp.x = __shfl_xor(t.x, 8, 32);
  tp.y = __shfl_xor(t.y, 8, 32);
  tp.z = __shfl_xor(t.z, 8, 32);
  tp.w = __shfl_xor(t.w, 8, 32);
  float4 o = t;
  if (lane < 16) {
    float sgn = (lane < 8) ? -1.f : 1.f;
    int p0 = lane * 4;
    #pragma unroll
    for (int e = 0; e < 4; ++e) {
      float fr = freqRow[p0 + e];
      float c = __cosf(fr), s = __sinf(fr);
      (&o.x)[e] = (&t.x)[e] * c + sgn * (&tp.x)[e] * s;
    }
  }
  return o;
}

__global__ __launch_bounds__(256) void k_qkv_post(
    const float* __restrict__ qproj, const float* __restrict__ kproj,
    const float* __restrict__ vproj, const float* __restrict__ freqs,
    const float* __restrict__ q_scale, const float* __restrict__ k_scale,
    const float* __restrict__ mem_k, const float* __restrict__ mem_v,
    unsigned short* __restrict__ qb, unsigned short* __restrict__ kb,
    unsigned short* __restrict__ vT) {
  int lane = threadIdx.x & 31, wave = threadIdx.x >> 5;
  int row = blockIdx.x * 8 + wave;  // [0, B*H*2064)
  int jrow = row % 2064;
  int h = (row / 2064) & 15;
  int bb = row / (2064 * 16);
  int d0 = lane * 4;

  if (jrow < 16) {
    float4 kv = *(const float4*)(mem_k + ((size_t)h * 16 + jrow) * 128 + d0);
    float ss = wave_sum32(kv.x * kv.x + kv.y * kv.y + kv.z * kv.z + kv.w * kv.w);
    float inv = 1.f / fmaxf(sqrtf(ss), 1e-12f);
    float4 ks;
    #pragma unroll
    for (int e = 0; e < 4; ++e) (&ks.x)[e] = (&kv.x)[e] * inv * k_scale[d0 + e];
    size_t ko = (((size_t)bb * 16 + h) * 2064 + jrow) * 128 + d0;
    *(ushort4*)(kb + ko) = pack4(ks);
    float4 vv = *(const float4*)(mem_v + ((size_t)h * 16 + jrow) * 128 + d0);
    #pragma unroll
    for (int e = 0; e < 4; ++e)
      vT[(((size_t)bb * 16 + h) * 128 + d0 + e) * 2064 + jrow] = f2bf((&vv.x)[e]);
    return;
  }

  int qi = jrow - 16;
  size_t base = ((size_t)bb * 2048 + qi) * 2048 + (size_t)h * 128 + d0;
  float4 q4 = *(const float4*)(qproj + base);
  float4 k4 = *(const float4*)(kproj + base);
  float4 v4 = *(const float4*)(vproj + base);
  float qss = wave_sum32(q4.x * q4.x + q4.y * q4.y + q4.z * q4.z + q4.w * q4.w);
  float kss = wave_sum32(k4.x * k4.x + k4.y * k4.y + k4.z * k4.z + k4.w * k4.w);
  float qin = 1.f / fmaxf(sqrtf(qss), 1e-12f);
  float kin = 1.f / fmaxf(sqrtf(kss), 1e-12f);
  float4 qs, ks;
  #pragma unroll
  for (int e = 0; e < 4; ++e) {
    (&qs.x)[e] = (&q4.x)[e] * qin * q_scale[d0 + e];
    (&ks.x)[e] = (&k4.x)[e] * kin * k_scale[d0 + e];
  }
  const float* freqRow = freqs + (size_t)qi * 64;
  qs = rope4(qs, freqRow, lane);
  ks = rope4(ks, freqRow, lane);
  float4 vs = rope4(v4, freqRow, lane);

  size_t qo = (((size_t)bb * 16 + h) * 2048 + qi) * 128 + d0;
  *(ushort4*)(qb + qo) = pack4(qs);
  size_t ko = (((size_t)bb * 16 + h) * 2064 + jrow) * 128 + d0;
  *(ushort4*)(kb + ko) = pack4(ks);
  #pragma unroll
  for (int e = 0; e < 4; ++e)
    vT[(((size_t)bb * 16 + h) * 128 + d0 + e) * 2064 + jrow] = f2bf((&vs.x)[e]);
}

// ---------------------------------------------------------------- flash attention
// Block = (batch, 16 query rows). K tiles for all 16 heads arrive via one
// Tensor Data Mover descriptor per key block (TENSORcnt); pre-talk mixing
// through LDS; online softmax; P@V with WMMA.
__global__ __launch_bounds__(256) void k_attention(
    const unsigned short* __restrict__ qb, const unsigned short* __restrict__ kb,
    const unsigned short* __restrict__ vT, const float* __restrict__ pre_talk,
    float* __restrict__ O) {
  __shared__ unsigned short Kbuf[16][32][128];  // 128 KB: [h][key][d]
  __shared__ float Sbuf[16][16][32];            // 32 KB raw scores
  __shared__ unsigned short Pbuf[16][16][32];   // 16 KB probs (bf16)
  __shared__ float preS[256];

  int tid = threadIdx.x;
  int lane = tid & 31, wave = tid >> 5;
  int lo = lane & 15, grp = lane >> 4;
  int qi0 = (blockIdx.x & 127) * 16;
  int bb = blockIdx.x >> 7;

  preS[tid] = pre_talk[tid] * 10.0f;  // fold QK_SCALE into pre_talk
  __syncthreads();

  float pw[2][16];
  #pragma unroll
  for (int g = 0; g < 2; ++g)
    #pragma unroll
    for (int hh = 0; hh < 16; ++hh) pw[g][hh] = preS[(2 * wave + g) * 16 + hh];

  v8f z = {0.f, 0.f, 0.f, 0.f, 0.f, 0.f, 0.f, 0.f};
  v8f accO[2][8];
  float mrun[2][8], lrun[2][8];
  #pragma unroll
  for (int g = 0; g < 2; ++g) {
    #pragma unroll
    for (int nt = 0; nt < 8; ++nt) accO[g][nt] = z;
    #pragma unroll
    for (int e = 0; e < 8; ++e) { mrun[g][e] = -3.0e38f; lrun[g][e] = 0.f; }
  }

  unsigned ldsK = (unsigned)(size_t)(void*)&Kbuf[0][0][0];

  int nblk = (qi0 + 63) >> 5;  // keys needed: qi0+32 (causal + 16 mem slots)
  if (nblk > 65) nblk = 65;

  for (int blk = 0; blk < nblk; ++blk) {
    int kb0 = blk * 32;

    // ---- TDM: K tile [16 heads][32 keys][128 d] -> LDS ----
    if (wave == 0) {
      unsigned long long ga = (unsigned long long)(uintptr_t)(
          kb + ((size_t)bb * 16 * 2064 + kb0) * 128);
      unsigned dim1 = (unsigned)(2064 - kb0);     // rows left; OOB keys -> 0
      v4u g0 = {1u,                                // count=1 descriptor
                ldsK,                              // lds_addr
                (unsigned)ga,
                (unsigned)((ga >> 32) & 0x1FFFFFFu) | 0x80000000u};  // type=2
      v8i g1 = {(int)(1u << 16),                   // data_size=2 bytes
                (int)(128u << 16),                 // tensor_dim0 lo16 (=128)
                (int)((dim1 & 0xFFFFu) << 16),     // dim0 hi | dim1 lo
                (int)(((dim1 >> 16) & 0xFFFFu) | (128u << 16)),  // tile_dim0=128
                (int)(32u | (16u << 16)),          // tile_dim1=32, tile_dim2=16
                (int)128u,                         // tensor_dim0_stride = 128
                (int)((264192u & 0xFFFFu) << 16),  // dim1_stride lo16 (2064*128)
                (int)(264192u >> 16)};             // dim1_stride hi
      v4i g2 = {16, 0, 0, 0};                      // tensor_dim2 = 16 heads
      v4i g3 = {0, 0, 0, 0};
#if __clang_major__ >= 23
      v8i gx = {0, 0, 0, 0, 0, 0, 0, 0};
      __builtin_amdgcn_tensor_load_to_lds(g0, g1, g2, g3, gx, 0);
#else
      __builtin_amdgcn_tensor_load_to_lds(g0, g1, g2, g3, 0);
#endif
      __builtin_amdgcn_s_wait_tensorcnt(0);
    }
    __syncthreads();

    // ---- QK^T for this wave's 2 raw heads (K from LDS) ----
    #pragma unroll
    for (int hh = 0; hh < 2; ++hh) {
      int habs = wave * 2 + hh;
      const unsigned short* qbase =
          qb + (((size_t)bb * 16 + habs) * 2048 + qi0 + lo) * 128;
      const unsigned short* kst = &Kbuf[habs][0][0];
      v8f s0 = z, s1 = z;
      #pragma unroll
      for (int dc = 0; dc < 4; ++dc) {
        v16bf aQ = load_fragA(qbase + dc * 32, lane);
        v16bf b0 = load_fragB(kst + (size_t)lo * 128 + dc * 32, lane);
        v16bf b1 = load_fragB(kst + (size_t)(16 + lo) * 128 + dc * 32, lane);
        s0 = wmma_bf16(aQ, b0, s0);
        s1 = wmma_bf16(aQ, b1, s1);
      }
      #pragma unroll
      for (int e = 0; e < 8; ++e) {
        Sbuf[habs][grp * 8 + e][lo] = s0[e];
        Sbuf[habs][grp * 8 + e][16 + lo] = s1[e];
      }
    }
    __syncthreads();

    // ---- pre-talk mix + causal mask ----
    float sp[2][2][8];
    #pragma unroll
    for (int jt = 0; jt < 2; ++jt) {
      int jg = kb0 + jt * 16 + lo;
      #pragma unroll
      for (int e = 0; e < 8; ++e) {
        int qi = qi0 + grp * 8 + e;
        float a0 = 0.f, a1 = 0.f;
        #pragma unroll
        for (int hh = 0; hh < 16; ++hh) {
          float sv = Sbuf[hh][grp * 8 + e][jt * 16 + lo];
          a0 += pw[0][hh] * sv;
          a1 += pw[1][hh] * sv;
        }
        bool ok = (jg <= qi + 16) && (jg < 2064);
        sp[0][jt][e] = ok ? a0 : -3.0e38f;
        sp[1][jt][e] = ok ? a1 : -3.0e38f;
      }
    }

    // ---- online softmax (row = half-wave of 16 lanes) ----
    #pragma unroll
    for (int g = 0; g < 2; ++g) {
      int gabs = wave * 2 + g;
      #pragma unroll
      for (int e = 0; e < 8; ++e) {
        float mb = fmaxf(sp[g][0][e], sp[g][1][e]);
        #pragma unroll
        for (int off = 1; off < 16; off <<= 1)
          mb = fmaxf(mb, __shfl_xor(mb, off, 16));
        float mnew = fmaxf(mrun[g][e], mb);
        float fsc = __expf(mrun[g][e] - mnew);
        float p0 = __expf(sp[g][0][e] - mnew);
        float p1 = __expf(sp[g][1][e] - mnew);
        float ls = p0 + p1;
        #pragma unroll
        for (int off = 1; off < 16; off <<= 1) ls += __shfl_xor(ls, off, 16);
        lrun[g][e] = lrun[g][e] * fsc + ls;
        mrun[g][e] = mnew;
        #pragma unroll
        for (int nt = 0; nt < 8; ++nt) accO[g][nt][e] *= fsc;
        Pbuf[gabs][grp * 8 + e][lo] = f2bf(p0);
        Pbuf[gabs][grp * 8 + e][16 + lo] = f2bf(p1);
      }
    }
    __syncthreads();

    // ---- P @ V (V^T global -> contiguous B-fragments) ----
    #pragma unroll
    for (int g = 0; g < 2; ++g) {
      int gabs = wave * 2 + g;
      v16bf aP = load_fragA(&Pbuf[gabs][lo][0], lane);
      #pragma unroll
      for (int nt = 0; nt < 8; ++nt) {
        const unsigned short* vbase =
            vT + (((size_t)bb * 16 + gabs) * 128 + nt * 16 + lo) * 2064 + kb0;
        v16bf bV = load_fragB(vbase, lane);
        accO[g][nt] = wmma_bf16(aP, bV, accO[g][nt]);
      }
    }
    __syncthreads();
  }

  // ---- normalize, write per-head outputs ----
  #pragma unroll
  for (int g = 0; g < 2; ++g) {
    int gabs = wave * 2 + g;
    #pragma unroll
    for (int e = 0; e < 8; ++e) {
      float l = lrun[g][e];
      float inv = (l > 0.f) ? 1.f / l : 0.f;
      int qi = qi0 + grp * 8 + e;
      #pragma unroll
      for (int nt = 0; nt < 8; ++nt)
        O[(((size_t)bb * 16 + gabs) * 2048 + qi) * 128 + nt * 16 + lo] =
            accO[g][nt][e] * inv;
    }
  }
}

// ---------------------------------------------------------------- post-talk + gate
__global__ __launch_bounds__(256) void k_postmix(
    const float* __restrict__ O, const float* __restrict__ post_talk,
    const float* __restrict__ head_scale, const float* __restrict__ gproj,
    const float* __restrict__ bg, unsigned short* __restrict__ gated) {
  int i = blockIdx.x * 256 + threadIdx.x;  // over B*N*2048
  if (i >= 2 * 2048 * 2048) return;
  int dfull = i & 2047;
  int bn = i >> 11;
  int bb = bn >> 11;
  int qi = bn & 2047;
  int g = dfull >> 7, d0 = dfull & 127;
  float s = 0.f;
  #pragma unroll
  for (int hh = 0; hh < 16; ++hh)
    s += post_talk[g * 16 + hh] *
         O[(((size_t)bb * 16 + hh) * 2048 + qi) * 128 + d0];
  float gate = gproj[(size_t)bn * 2048 + dfull] + bg[dfull];
  gate = 1.f / (1.f + __expf(-gate));
  gated[(size_t)bn * 2048 + dfull] = f2bf(head_scale[g] * s * gate);
}

// ================================================================ launcher
extern "C" void kernel_launch(void* const* d_in, const int* in_sizes, int n_in,
                              void* d_out, int out_size, void* d_ws, size_t ws_size,
                              hipStream_t stream) {
  (void)in_sizes; (void)n_in; (void)out_size; (void)ws_size;
  const float* x        = (const float*)d_in[0];
  const float* freqs    = (const float*)d_in[2];
  const float* Wq       = (const float*)d_in[3];
  const float* Wk       = (const float*)d_in[4];
  const float* Wv       = (const float*)d_in[5];
  const float* q_scale  = (const float*)d_in[6];
  const float* k_scale  = (const float*)d_in[7];
  const float* mem_k    = (const float*)d_in[8];
  const float* mem_v    = (const float*)d_in[9];
  const float* pre_talk = (const float*)d_in[10];
  const float* post_talk= (const float*)d_in[11];
  const float* hscale   = (const float*)d_in[12];
  const float* Wg       = (const float*)d_in[13];
  const float* bg       = (const float*)d_in[14];
  const float* Wo       = (const float*)d_in[15];

  const size_t BN = 4096, HD = 2048;
  size_t off = 0;
  char* ws = (char*)d_ws;
  auto alloc = [&](size_t bytes) -> void* {
    void* p = ws + off;
    off += (bytes + 255) & ~(size_t)255;
    return p;
  };
  unsigned short* XB  = (unsigned short*)alloc(BN * HD * 2);
  unsigned short* WQT = (unsigned short*)alloc(HD * HD * 2);
  unsigned short* WKT = (unsigned short*)alloc(HD * HD * 2);
  unsigned short* WVT = (unsigned short*)alloc(HD * HD * 2);
  unsigned short* WGT = (unsigned short*)alloc(HD * HD * 2);
  unsigned short* WOT = (unsigned short*)alloc(HD * HD * 2);
  float* QP = (float*)alloc(BN * HD * 4);
  float* KP = (float*)alloc(BN * HD * 4);
  float* VP = (float*)alloc(BN * HD * 4);
  float* GP = (float*)alloc(BN * HD * 4);
  unsigned short* QB = (unsigned short*)alloc((size_t)2 * 16 * 2048 * 128 * 2);
  unsigned short* KB = (unsigned short*)alloc(((size_t)2 * 16 * 2064 * 128 + 8192) * 2);
  unsigned short* VT = (unsigned short*)alloc(((size_t)2 * 16 * 2064 * 128 + 8192) * 2);
  float* O = QP;                                // reuse: qproj dead after qkv_post
  unsigned short* GATED = (unsigned short*)KP;  // reuse: kproj dead after qkv_post

  k_cvt_bf16<<<(int)((BN * HD + 255) / 256), 256, 0, stream>>>(x, XB, (int)(BN * HD));
  int tB = (int)((HD * HD + 255) / 256);
  k_transpose_cvt<<<tB, 256, 0, stream>>>(Wq, WQT, 2048, 2048);
  k_transpose_cvt<<<tB, 256, 0, stream>>>(Wk, WKT, 2048, 2048);
  k_transpose_cvt<<<tB, 256, 0, stream>>>(Wv, WVT, 2048, 2048);
  k_transpose_cvt<<<tB, 256, 0, stream>>>(Wg, WGT, 2048, 2048);
  k_transpose_cvt<<<tB, 256, 0, stream>>>(Wo, WOT, 2048, 2048);

  dim3 gg(2048 / 128, 4096 / 128);
  k_gemm_bf16<<<gg, 256, 0, stream>>>(XB, WQT, QP, 4096, 2048, 2048);
  k_gemm_bf16<<<gg, 256, 0, stream>>>(XB, WKT, KP, 4096, 2048, 2048);
  k_gemm_bf16<<<gg, 256, 0, stream>>>(XB, WVT, VP, 4096, 2048, 2048);
  k_gemm_bf16<<<gg, 256, 0, stream>>>(XB, WGT, GP, 4096, 2048, 2048);

  k_qkv_post<<<(2 * 16 * 2064) / 8, 256, 0, stream>>>(
      QP, KP, VP, freqs, q_scale, k_scale, mem_k, mem_v, QB, KB, VT);

  k_attention<<<2 * 128, 256, 0, stream>>>(QB, KB, VT, pre_talk, O);

  k_postmix<<<(int)((BN * HD + 255) / 256), 256, 0, stream>>>(
      O, post_talk, hscale, GP, bg, GATED);

  k_gemm_bf16<<<gg, 256, 0, stream>>>(GATED, WOT, (float*)d_out, 4096, 2048, 2048);
}
